// BertSelfAttention_36704790512144
// MI455X (gfx1250) — matrix operand order repaired and checked by reading, not compiled, not run on hardware
//
#include <hip/hip_runtime.h>
#include <hip/hip_bf16.h>

#define DMODEL 1024
#define NHEAD  16
#define DK     64
#define BATCH  2
#define SEQ    2048
#define MTOK   (BATCH * SEQ)   // 4096 tokens

typedef __attribute__((ext_vector_type(16))) __bf16 v16bf;
typedef __attribute__((ext_vector_type(8)))  __bf16 v8bf;
typedef __attribute__((ext_vector_type(8)))  float  v8f;

// ---------------- WMMA helpers (CDNA5 bf16 16x16x32, f32 acc) ----------------

__device__ __forceinline__ v8f wmma_bf16(v16bf a, v16bf b, v8f c) {
    return __builtin_amdgcn_wmma_f32_16x16x32_bf16(false, a, false, b, (short)0, c, false, false);
}

// A fragment, 16x32 (MxK): lane lo holds row lo.
// elem e<8  -> k = e + 8*hi ; elem e>=8 -> k = 16 + 8*hi + (e-8)
__device__ __forceinline__ v16bf frag_A(const __bf16* rowp, int hi) {
    v8bf a = *(const v8bf*)(rowp + 8 * hi);
    v8bf b = *(const v8bf*)(rowp + 16 + 8 * hi);
    v16bf f;
#pragma unroll
    for (int i = 0; i < 8; ++i) { f[i] = a[i]; f[i + 8] = b[i]; }
    return f;
}

// B fragment, 32x16 (KxN): lane lo holds column n=lo; elem e<8 -> k = e + 8*hi ; elem e>=8 -> k = 16 + 8*hi + (e-8).
// rowp points at Bt[n] + kBase (Bt row-major [N][K]).
__device__ __forceinline__ v16bf frag_B(const __bf16* rowp, int hi) {
    v8bf a = *(const v8bf*)(rowp + 8 * hi);
    v8bf b = *(const v8bf*)(rowp + 16 + 8 * hi);
    v16bf f;
#pragma unroll
    for (int i = 0; i < 8; ++i) { f[i] = a[i]; f[i + 8] = b[i]; }
    return f;
}

// ---------------- conversion kernels ----------------

__global__ __launch_bounds__(256) void cvt_x_bf16(const float* __restrict__ x,
                                                  __bf16* __restrict__ xb) {
    int i = (blockIdx.x * 256 + threadIdx.x) * 4;
#pragma unroll
    for (int r = 0; r < 4; ++r) xb[i + r] = (__bf16)x[i + r];
}

// WT[j][d] = W[d][j], fp32 -> bf16
__global__ __launch_bounds__(256) void transpose_w_bf16(const float* __restrict__ W,
                                                        __bf16* __restrict__ WT) {
    __shared__ float t[32][33];
    int bx = blockIdx.x * 32, by = blockIdx.y * 32;
    int x = threadIdx.x & 31, y = threadIdx.x >> 5;
#pragma unroll
    for (int r = y; r < 32; r += 8) t[r][x] = W[(size_t)(by + r) * DMODEL + bx + x];
    __syncthreads();
#pragma unroll
    for (int r = y; r < 32; r += 8) WT[(size_t)(bx + r) * DMODEL + by + x] = (__bf16)t[x][r];
}

// ------------- async stage: 64 cols x 32 k of W^T into LDS (one b128 per thread) -------------
#define LDSB_STRIDE 40   // halfs per row (80 B: 16B aligned, conflict-breaking pad)

__device__ __forceinline__ void stage_wt_async(const __bf16* __restrict__ WT, int col0, int kc,
                                               __bf16* ldsbuf, int tid) {
    int row = tid >> 2;                       // 0..63  (column of W)
    int seg = tid & 3;                        // 16B segment within the 32-half k slice
    const __bf16* g = WT + (size_t)(col0 + row) * DMODEL + 32 * kc + seg * 8;
    unsigned lofs = (unsigned)(unsigned long long)(void*)(ldsbuf + row * LDSB_STRIDE + seg * 8);
    asm volatile("global_load_async_to_lds_b128 %0, %1, off"
                 :: "v"(lofs), "v"(g) : "memory");
}

// ---------------- fused QKV projection GEMM ----------------
// Block: 8 waves share one (mat, col group); each wave owns a 16-row token tile.
__global__ __launch_bounds__(256) void gemm_qkv(
    const __bf16* __restrict__ Xb,
    const __bf16* __restrict__ WqT, const __bf16* __restrict__ WkT, const __bf16* __restrict__ WvT,
    const float* __restrict__ bq, const float* __restrict__ bk, const float* __restrict__ bv,
    __bf16* __restrict__ Qb, __bf16* __restrict__ Kb, __bf16* __restrict__ Vtb) {
    __shared__ __align__(16) __bf16 bt[2][64 * LDSB_STRIDE];

    int tid = threadIdx.x;
    int lane = tid & 31, wave = tid >> 5;
    int lo = lane & 15, hi = lane >> 4;
    int blk = blockIdx.x;                      // 0..1535
    int mat = blk >> 9;                        // /512
    int r = blk & 511;
    int rowBlk = r >> 4;                       // 0..31
    int col0 = (r & 15) * 64;
    int row0 = rowBlk * 128 + wave * 16;

    const __bf16* WT   = (mat == 0) ? WqT : (mat == 1) ? WkT : WvT;
    const float*  bias = (mat == 0) ? bq  : (mat == 1) ? bk  : bv;
    const __bf16* arow = Xb + (size_t)(row0 + lo) * DMODEL;

    stage_wt_async(WT, col0, 0, bt[0], tid);
    v16bf a = frag_A(arow, hi);
    v8f acc[4] = {};
    for (int c = 0; c < DMODEL / 32; ++c) {
        v16bf an;
        if (c < DMODEL / 32 - 1) {
            stage_wt_async(WT, col0, c + 1, bt[(c + 1) & 1], tid);   // overlap next DMA
            an = frag_A(arow + 32 * (c + 1), hi);
            asm volatile("s_wait_asynccnt 0x1" ::: "memory");        // current tile landed
        } else {
            asm volatile("s_wait_asynccnt 0x0" ::: "memory");
        }
        __syncthreads();
        const __bf16* lb = bt[c & 1];
#pragma unroll
        for (int nt = 0; nt < 4; ++nt) {
            v16bf b = frag_B(lb + (16 * nt + lo) * LDSB_STRIDE, hi);
            acc[nt] = wmma_bf16(a, b, acc[nt]);
        }
        __syncthreads();
        a = an;
    }
#pragma unroll
    for (int nt = 0; nt < 4; ++nt) {
#pragma unroll
        for (int e = 0; e < 8; ++e) {
            int i = row0 + e + 8 * hi;                 // token
            int j = col0 + 16 * nt + lo;               // feature
            float v = acc[nt][e] + bias[j];
            int bb = i >> 11, s = i & (SEQ - 1);
            int hh = j >> 6,  d = j & (DK - 1);
            if (mat == 2) {
                Vtb[(((size_t)bb * NHEAD + hh) * DK + d) * SEQ + s] = (__bf16)v;
            } else {
                float sv = (mat == 0) ? v * 0.125f : v;   // fold 1/sqrt(dk) into Q
                __bf16* dst = (mat == 0) ? Qb : Kb;
                dst[(((size_t)bb * NHEAD + hh) * SEQ + s) * DK + d] = (__bf16)sv;
            }
        }
    }
}

// ---------------- attention: scores + softmax + PV, one 16-row strip per wave ----------------
__global__ __launch_bounds__(256) void attn_kernel(
    const __bf16* __restrict__ Qb, const __bf16* __restrict__ Kb, const __bf16* __restrict__ Vtb,
    float* __restrict__ attnW, __bf16* __restrict__ AOb) {
    __shared__ __align__(16) __bf16 ptile[8][16][32];

    int lane = threadIdx.x & 31, wave = threadIdx.x >> 5;
    int lo = lane & 15, hi = lane >> 4;
    int strip = blockIdx.x * 8 + wave;        // 0..4095
    int bh = strip >> 7;
    int row0 = (strip & 127) * 16;
    int b = bh >> 4, h = bh & 15;

    const __bf16* qrow = Qb + ((size_t)bh * SEQ + row0 + lo) * DK;
    v16bf qa0 = frag_A(qrow, hi);
    v16bf qa1 = frag_A(qrow + 32, hi);

    const __bf16* kbase = Kb + (size_t)bh * SEQ * DK;
    float* sc = attnW + ((size_t)bh * SEQ + row0) * SEQ;

    float pmax[8], psum[8];
#pragma unroll
    for (int e = 0; e < 8; ++e) { pmax[e] = -1.0e30f; psum[e] = 0.0f; }

    // ---- pass 1: raw scores + per-lane online max/sum (K frags pipelined) ----
    v16bf kb0 = frag_B(kbase + (size_t)lo * DK, hi);
    v16bf kb1 = frag_B(kbase + (size_t)lo * DK + 32, hi);
    for (int j = 0; j < SEQ / 16; ++j) {
        v16bf nb0, nb1;
        if (j < SEQ / 16 - 1) {
            const __bf16* kr = kbase + (size_t)((j + 1) * 16 + lo) * DK;
            nb0 = frag_B(kr, hi);
            nb1 = frag_B(kr + 32, hi);
        }
        v8f acc = {};
        acc = wmma_bf16(qa0, kb0, acc);
        acc = wmma_bf16(qa1, kb1, acc);
#pragma unroll
        for (int e = 0; e < 8; ++e) {
            float s = acc[e];
            sc[(size_t)(e + 8 * hi) * SEQ + j * 16 + lo] = s;
            float nm = fmaxf(pmax[e], s);
            psum[e] = psum[e] * __expf(pmax[e] - nm) + __expf(s - nm);
            pmax[e] = nm;
        }
        kb0 = nb0; kb1 = nb1;
    }
#pragma unroll
    for (int off = 1; off < 16; off <<= 1) {
#pragma unroll
        for (int e = 0; e < 8; ++e) {
            float om = __shfl_xor(pmax[e], off);
            float os = __shfl_xor(psum[e], off);
            float nm = fmaxf(pmax[e], om);
            psum[e] = psum[e] * __expf(pmax[e] - nm) + os * __expf(om - nm);
            pmax[e] = nm;
        }
    }
    float inv[8];
#pragma unroll
    for (int e = 0; e < 8; ++e) inv[e] = 1.0f / psum[e];

    // ---- pass 2: normalize (rewrite attn) + P@V (V frags issued before LDS round-trip) ----
    const __bf16* vbase = Vtb + (size_t)bh * DK * SEQ;
    v8f oacc[4] = {};
    for (int c = 0; c < SEQ / 32; ++c) {
        v16bf vb[4];
#pragma unroll
        for (int nt = 0; nt < 4; ++nt)
            vb[nt] = frag_B(vbase + (size_t)(16 * nt + lo) * SEQ + 32 * c, hi);
#pragma unroll
        for (int half = 0; half < 2; ++half) {
            int j = 2 * c + half;
#pragma unroll
            for (int e = 0; e < 8; ++e) {
                size_t idx = (size_t)(e + 8 * hi) * SEQ + j * 16 + lo;
                float p = __expf(sc[idx] - pmax[e]) * inv[e];
                sc[idx] = p;
                ptile[wave][e + 8 * hi][16 * half + lo] = (__bf16)p;
            }
        }
        asm volatile("" ::: "memory");   // LDS in-order per wave; fence the compiler only
        v16bf pa = frag_A(&ptile[wave][lo][0], hi);
#pragma unroll
        for (int nt = 0; nt < 4; ++nt) oacc[nt] = wmma_bf16(pa, vb[nt], oacc[nt]);
        asm volatile("" ::: "memory");
    }
#pragma unroll
    for (int nt = 0; nt < 4; ++nt) {
#pragma unroll
        for (int e = 0; e < 8; ++e) {
            int m = e + 8 * hi;
            int d = 16 * nt + lo;
            AOb[((size_t)b * SEQ + row0 + m) * DMODEL + h * DK + d] = (__bf16)oacc[nt][e];
        }
    }
}

// ---------------- output projection GEMM -> fp32 d_out ----------------
__global__ __launch_bounds__(256) void gemm_out(
    const __bf16* __restrict__ AOb, const __bf16* __restrict__ WoT,
    const float* __restrict__ bo, float* __restrict__ out) {
    __shared__ __align__(16) __bf16 bt[2][64 * LDSB_STRIDE];

    int tid = threadIdx.x;
    int lane = tid & 31, wave = tid >> 5;
    int lo = lane & 15, hi = lane >> 4;
    int blk = blockIdx.x;                      // 0..511
    int rowBlk = blk >> 4;                     // 0..31
    int col0 = (blk & 15) * 64;
    int row0 = rowBlk * 128 + wave * 16;

    const __bf16* arow = AOb + (size_t)(row0 + lo) * DMODEL;
    stage_wt_async(WoT, col0, 0, bt[0], tid);
    v16bf a = frag_A(arow, hi);
    v8f acc[4] = {};
    for (int c = 0; c < DMODEL / 32; ++c) {
        v16bf an;
        if (c < DMODEL / 32 - 1) {
            stage_wt_async(WoT, col0, c + 1, bt[(c + 1) & 1], tid);
            an = frag_A(arow + 32 * (c + 1), hi);
            asm volatile("s_wait_asynccnt 0x1" ::: "memory");
        } else {
            asm volatile("s_wait_asynccnt 0x0" ::: "memory");
        }
        __syncthreads();
        const __bf16* lb = bt[c & 1];
#pragma unroll
        for (int nt = 0; nt < 4; ++nt) {
            v16bf b = frag_B(lb + (16 * nt + lo) * LDSB_STRIDE, hi);
            acc[nt] = wmma_bf16(a, b, acc[nt]);
        }
        __syncthreads();
        a = an;
    }
#pragma unroll
    for (int nt = 0; nt < 4; ++nt) {
#pragma unroll
        for (int e = 0; e < 8; ++e) {
            int i = row0 + e + 8 * hi;
            int j = col0 + 16 * nt + lo;
            out[(size_t)i * DMODEL + j] = acc[nt][e] + bo[j];
        }
    }
}

// ---------------- launcher ----------------
extern "C" void kernel_launch(void* const* d_in, const int* in_sizes, int n_in,
                              void* d_out, int out_size, void* d_ws, size_t ws_size,
                              hipStream_t stream) {
    const float* x  = (const float*)d_in[0];
    const float* wq = (const float*)d_in[1];
    const float* bq = (const float*)d_in[2];
    const float* wk = (const float*)d_in[3];
    const float* bk = (const float*)d_in[4];
    const float* wv = (const float*)d_in[5];
    const float* bv = (const float*)d_in[6];
    const float* wo = (const float*)d_in[7];
    const float* bo = (const float*)d_in[8];

    char* ws = (char*)d_ws;
    size_t off = 0;
    __bf16* Xb  = (__bf16*)(ws + off); off += (size_t)MTOK * DMODEL * 2;
    __bf16* WqT = (__bf16*)(ws + off); off += (size_t)DMODEL * DMODEL * 2;
    __bf16* WkT = (__bf16*)(ws + off); off += (size_t)DMODEL * DMODEL * 2;
    __bf16* WvT = (__bf16*)(ws + off); off += (size_t)DMODEL * DMODEL * 2;
    __bf16* WoT = (__bf16*)(ws + off); off += (size_t)DMODEL * DMODEL * 2;
    __bf16* Qb  = (__bf16*)(ws + off); off += (size_t)MTOK * DMODEL * 2;      // [B,H,S,dk]
    __bf16* Kb  = (__bf16*)(ws + off); off += (size_t)MTOK * DMODEL * 2;      // [B,H,S,dk]
    __bf16* Vtb = (__bf16*)(ws + off); off += (size_t)MTOK * DMODEL * 2;      // [B,H,dk,S]
    __bf16* AOb = (__bf16*)(ws + off); off += (size_t)MTOK * DMODEL * 2;      // [B,S,D]

    float* out   = (float*)d_out;
    float* attnW = (float*)d_out + (size_t)MTOK * DMODEL;

    cvt_x_bf16<<<(MTOK * DMODEL) / (256 * 4), 256, 0, stream>>>(x, Xb);
    dim3 tg(DMODEL / 32, DMODEL / 32);
    transpose_w_bf16<<<tg, 256, 0, stream>>>(wq, WqT);
    transpose_w_bf16<<<tg, 256, 0, stream>>>(wk, WkT);
    transpose_w_bf16<<<tg, 256, 0, stream>>>(wv, WvT);
    transpose_w_bf16<<<tg, 256, 0, stream>>>(wo, WoT);

    gemm_qkv<<<1536, 256, 0, stream>>>(Xb, WqT, WkT, WvT, bq, bk, bv, Qb, Kb, Vtb);
    attn_kernel<<<512, 256, 0, stream>>>(Qb, Kb, Vtb, attnW, AOb);
    gemm_out<<<512, 256, 0, stream>>>(AOb, WoT, bo, out);
}
